// CausalSelfAttention_86715389706427
// MI455X (gfx1250) — matrix operand order repaired and checked by reading, not compiled, not run on hardware
//
#include <hip/hip_runtime.h>

typedef __attribute__((ext_vector_type(16))) _Float16 v16h;
typedef __attribute__((ext_vector_type(8)))  _Float16 v8h;
typedef __attribute__((ext_vector_type(8)))  float    v8f;
typedef __attribute__((ext_vector_type(4)))  unsigned int v4u;
typedef __attribute__((ext_vector_type(8)))  int v8i;
typedef __attribute__((ext_vector_type(4)))  int v4i;

constexpr int DIN    = 1024;   // model dim
constexpr int DQKV   = 3072;   // 3 * d_out
constexpr int NHEAD  = 16;
constexpr int HDIM   = 64;
constexpr int SEQ    = 2048;
constexpr int MTOT   = 4096;   // NBATCH * SEQ
constexpr int BHN    = 32;     // NBATCH * NHEAD

#if __has_builtin(__builtin_amdgcn_tensor_load_to_lds) && __has_builtin(__builtin_amdgcn_s_wait_tensorcnt)
#define HAVE_TDM 1
#endif

// ---------------------------------------------------------------------------
// WMMA fragment loaders (CDNA5 wave32, 16-bit operands in this code's contiguous k order, same for A and B)
// A (16x32, MxK): lane m=lane&15; element e -> k = 16*(lane>>4) + e
// B (32x16, KxN) read from a [N][K] row-major buffer: lane n=lane&15;
//   element e -> k = 16*(lane>>4) + e  => one contiguous 16-half (32B) chunk.
// ---------------------------------------------------------------------------
__device__ __forceinline__ v16h load_frag_a(const _Float16* base, int ld, int lane) {
    const int m = lane & 15, hf = lane >> 4;
    const _Float16* p = base + (size_t)m * ld + hf * 16;
    union { v16h v; v8h h2[2]; } u;
    u.h2[0] = *(const v8h*)(p);
    u.h2[1] = *(const v8h*)(p + 8);
    return u.v;
}

__device__ __forceinline__ v16h load_frag_b(const _Float16* baseT, int ld, int lane) {
    const int n = lane & 15, hf = lane >> 4;
    return *(const v16h*)(baseT + (size_t)n * ld + hf * 16);
}

__device__ __forceinline__ v8f wmma_f16(v16h a, v16h b, v8f c) {
    return __builtin_amdgcn_wmma_f32_16x16x32_f16(false, a, false, b, (short)0, c, false, false);
}

// Cross-lane reductions over the 16-lane half-groups that share a C-row.
__device__ __forceinline__ float rowmax16(float v) {
    v = fmaxf(v, __shfl_xor(v, 1, 32));
    v = fmaxf(v, __shfl_xor(v, 2, 32));
    v = fmaxf(v, __shfl_xor(v, 4, 32));
    v = fmaxf(v, __shfl_xor(v, 8, 32));
    return v;
}
__device__ __forceinline__ float rowsum16(float v) {
    v += __shfl_xor(v, 1, 32);
    v += __shfl_xor(v, 2, 32);
    v += __shfl_xor(v, 4, 32);
    v += __shfl_xor(v, 8, 32);
    return v;
}

#ifdef HAVE_TDM
// Tensor Data Mover: stage a 256(row) x 32(col) f16 tile from a row-major
// [rows][1024] f16 matrix into LDS. D# layout per CDNA5 ISA:
//   group0: count=1 | lds_addr | global_addr | type=2
//   group1: data_size=1(2B); tensor_dim0=32, tensor_dim1=256 (tile-sized, no OOB);
//           tile_dim0=32, tile_dim1=256, tile_dim2=0; tensor_dim0_stride=1024.
__device__ __forceinline__ void tdm_stage_b(const _Float16* gsrc, unsigned lds_byte_off) {
    unsigned long long ga = (unsigned long long)(uintptr_t)gsrc;
    v4u g0 = { 1u,                                   // count=1, user descriptor
               lds_byte_off,                          // lds_addr
               (unsigned)ga,                          // global_addr[31:0]
               (unsigned)(ga >> 32) | 0x80000000u };  // global_addr[56:32] | type=2
    v8i g1 = { (int)0x00010000u,      // workgroup_mask=0, data_size=1 (2 bytes)
               (int)(32u << 16),      // tensor_dim0[15:0]=32 (bits 63:48)
               (int)(256u << 16),     // tensor_dim0[31:16]=0, tensor_dim1[15:0]=256
               (int)(32u << 16),      // tensor_dim1[31:16]=0, tile_dim0=32
               (int)256,              // tile_dim1=256, tile_dim2=0
               (int)1024,             // tensor_dim0_stride[31:0]=1024
               0,                     // stride0[47:32]=0, stride1[15:0]=0
               0 };                   // stride1[47:16]=0
    v4i z4 = {0, 0, 0, 0};
#if __clang_major__ >= 23
    v8i z8 = {0, 0, 0, 0, 0, 0, 0, 0};
    __builtin_amdgcn_tensor_load_to_lds(g0, g1, z4, z4, z8, 0);
#else
    __builtin_amdgcn_tensor_load_to_lds(g0, g1, z4, z4, 0);
#endif
}
#endif

// ---------------------------------------------------------------------------
// Conversion kernels
// ---------------------------------------------------------------------------
__global__ void k_cvt(const float* __restrict__ s, _Float16* __restrict__ d, int n) {
    int i = blockIdx.x * blockDim.x + threadIdx.x;
    if (i < n) d[i] = (_Float16)s[i];
}

// W [R][C] f32 -> Wt [C][R] f16
__global__ void k_cvt_t(const float* __restrict__ s, _Float16* __restrict__ d, int R, int C) {
    int i = blockIdx.x * blockDim.x + threadIdx.x;
    if (i < R * C) {
        int r = i / C, c = i - r * C;
        d[(size_t)c * R + r] = (_Float16)s[i];
    }
}

// ---------------------------------------------------------------------------
// QKV projection: [MTOT,DIN] @ [DIN,DQKV] + bias.
// Block: 8 waves, block tile 64(m) x 256(n); wave tile 32x64 (2x4 WMMA).
// B tile (256x32 f16, 16KB/step) staged to LDS by the TDM, double-buffered;
// wave 0 issues tensor_load_to_lds + s_wait_tensorcnt, block barrier publishes.
// A fragments load direct from global (4x reuse via WGP$).
// Epilogue scatters Q,K as [bh][t][hd] and V transposed as [bh][hd][t].
// ---------------------------------------------------------------------------
__global__ void __launch_bounds__(256) k_qkv_gemm(const _Float16* __restrict__ X,
                                                  const _Float16* __restrict__ Wt,
                                                  const float* __restrict__ bias,
                                                  _Float16* __restrict__ Q,
                                                  _Float16* __restrict__ K,
                                                  _Float16* __restrict__ Vt) {
    __shared__ __align__(32) _Float16 Bs[2][256 * 32];
    const int lane = threadIdx.x & 31;
    const int wave = threadIdx.x >> 5;
    const int m0 = blockIdx.x * 64 + (wave & 1) * 32;
    const int nblk = blockIdx.y * 256;
    const int wn = wave >> 1;                 // 0..3, 64 output cols each
    v8f acc[2][4] = {};

#ifdef HAVE_TDM
    if (wave == 0)
        tdm_stage_b(Wt + (size_t)nblk * DIN, (unsigned)(uintptr_t)&Bs[0][0]);
#endif
    for (int k0 = 0; k0 < DIN; k0 += 32) {
        const int buf = (k0 >> 5) & 1;
#ifdef HAVE_TDM
        if (wave == 0) __builtin_amdgcn_s_wait_tensorcnt(0);   // buf ready
        __syncthreads();                                       // publish to all waves
        if (wave == 0 && k0 + 32 < DIN)                        // prefetch next tile
            tdm_stage_b(Wt + (size_t)nblk * DIN + (k0 + 32),
                        (unsigned)(uintptr_t)&Bs[buf ^ 1][0]);
#else
        __syncthreads();   // previous iteration's readers done
        {                  // cooperative sync staging: 256 thr x 4 x 16B chunks
            const _Float16* src = Wt + (size_t)nblk * DIN + k0;
            for (int c = 0; c < 4; ++c) {
                int chunk = threadIdx.x * 4 + c;     // 0..1023, 8 halfs each
                int row = chunk >> 2, ko = (chunk & 3) * 8;
                *(v8h*)&Bs[buf][row * 32 + ko] = *(const v8h*)(src + (size_t)row * DIN + ko);
            }
        }
        __syncthreads();
#endif
        v16h a0 = load_frag_a(X + (size_t)m0 * DIN + k0, DIN, lane);
        v16h a1 = load_frag_a(X + (size_t)(m0 + 16) * DIN + k0, DIN, lane);
        for (int j = 0; j < 4; ++j) {
            v16h b = load_frag_b(&Bs[buf][(wn * 64 + j * 16) * 32], 32, lane);
            acc[0][j] = wmma_f16(a0, b, acc[0][j]);
            acc[1][j] = wmma_f16(a1, b, acc[1][j]);
        }
    }

    const int nn = lane & 15, hf = lane >> 4;
    for (int i = 0; i < 2; ++i)
        for (int j = 0; j < 4; ++j)
            for (int r = 0; r < 8; ++r) {
                int row = m0 + i * 16 + r + 8 * hf;           // 0..4095
                int col = nblk + wn * 64 + j * 16 + nn;       // 0..3071
                float v = acc[i][j][r] + bias[col];
                int b = row >> 11, t = row & (SEQ - 1);
                int sec = col >> 10, dd = col & 1023;
                int hh = dd >> 6, hd = dd & 63;
                size_t bh = (size_t)(b * NHEAD + hh);
                if (sec == 0)      Q[(bh * SEQ + t) * HDIM + hd] = (_Float16)v;
                else if (sec == 1) K[(bh * SEQ + t) * HDIM + hd] = (_Float16)v;
                else               Vt[(bh * HDIM + hd) * SEQ + t] = (_Float16)v;
            }
}

// ---------------------------------------------------------------------------
// Flash-style causal attention. One wave owns 16 queries of one (b,h) pair;
// streams keys in blocks of 32 with online softmax; P goes through per-wave
// LDS to re-layout C-matrix -> A-fragment for the PV WMMA.
// ---------------------------------------------------------------------------
__global__ void __launch_bounds__(256) k_attn(const _Float16* __restrict__ Q,
                                              const _Float16* __restrict__ K,
                                              const _Float16* __restrict__ Vt,
                                              _Float16* __restrict__ ctx) {
    __shared__ __align__(32) _Float16 pbuf[8][16 * 32];
    const int lane = threadIdx.x & 31;
    const int wave = threadIdx.x >> 5;
    const int bh = blockIdx.y;                 // 0..31
    const int q0 = (blockIdx.x * 8 + wave) * 16;
    const _Float16* Qb = Q + (size_t)bh * SEQ * HDIM;
    const _Float16* Kb = K + (size_t)bh * SEQ * HDIM;
    const _Float16* Vb = Vt + (size_t)bh * HDIM * SEQ;
    _Float16* pb = pbuf[wave];
    const int nn = lane & 15, hf = lane >> 4;

    float mrow[8], lrow[8];
    v8f acc[4] = {};
    for (int r = 0; r < 8; ++r) { mrow[r] = -1e30f; lrow[r] = 0.0f; }

    // Q fragments are loop-invariant: 16 rows x hd=64 -> two A fragments.
    v16h qa0 = load_frag_a(Qb + (size_t)q0 * HDIM + 0, HDIM, lane);
    v16h qa1 = load_frag_a(Qb + (size_t)q0 * HDIM + 32, HDIM, lane);

    for (int kb = 0; kb <= q0 + 15; kb += 32) {
        // prefetch next key block into cache while we compute this one
        __builtin_prefetch((const void*)(Kb + (size_t)(kb + 32) * HDIM), 0, 3);
        // scores: 16 x 32 keys (two 16x16 C tiles), K=hd=64 in two steps
        v8f s0 = {}, s1 = {};
        {
            v16h f;
            f = load_frag_b(Kb + (size_t)kb * HDIM + 0, HDIM, lane);         s0 = wmma_f16(qa0, f, s0);
            f = load_frag_b(Kb + (size_t)kb * HDIM + 32, HDIM, lane);        s0 = wmma_f16(qa1, f, s0);
            f = load_frag_b(Kb + (size_t)(kb + 16) * HDIM + 0, HDIM, lane);  s1 = wmma_f16(qa0, f, s1);
            f = load_frag_b(Kb + (size_t)(kb + 16) * HDIM + 32, HDIM, lane); s1 = wmma_f16(qa1, f, s1);
        }
        float sv0[8], sv1[8];
        for (int r = 0; r < 8; ++r) {
            int tq = q0 + r + 8 * hf;
            sv0[r] = ((kb + nn) > tq)      ? -1e30f : s0[r] * 0.125f;  // 1/sqrt(64)
            sv1[r] = ((kb + 16 + nn) > tq) ? -1e30f : s1[r] * 0.125f;
        }
        for (int r = 0; r < 8; ++r) {
            float rmax = rowmax16(fmaxf(sv0[r], sv1[r]));
            float mnew = fmaxf(mrow[r], rmax);
            float corr = __expf(mrow[r] - mnew);
            mrow[r] = mnew;
            float p0 = __expf(sv0[r] - mnew);
            float p1 = __expf(sv1[r] - mnew);
            sv0[r] = p0; sv1[r] = p1;
            lrow[r] = lrow[r] * corr + rowsum16(p0 + p1);
            for (int n = 0; n < 4; ++n) acc[n][r] *= corr;
        }
        // C-layout -> A-fragment re-layout via per-wave LDS tile (16 rows x 32 keys)
        for (int r = 0; r < 8; ++r) {
            pb[(r + 8 * hf) * 32 + nn]      = (_Float16)sv0[r];
            pb[(r + 8 * hf) * 32 + 16 + nn] = (_Float16)sv1[r];
        }
        asm volatile("s_wait_dscnt 0" ::: "memory");   // LDS RAW within the wave
        v16h pa = load_frag_a(pb, 32, lane);
        for (int n = 0; n < 4; ++n) {
            v16h vb = load_frag_b(Vb + (size_t)(n * 16) * SEQ + kb, SEQ, lane);
            acc[n] = wmma_f16(pa, vb, acc[n]);
        }
        asm volatile("" ::: "memory");  // keep next iter's LDS stores after these loads
    }

    const int hh = bh & 15, bb = bh >> 4;
    for (int n = 0; n < 4; ++n)
        for (int r = 0; r < 8; ++r) {
            int t = q0 + r + 8 * hf;
            float o = acc[n][r] / lrow[r];
            ctx[((size_t)(bb * SEQ + t)) * DIN + hh * HDIM + n * 16 + nn] = (_Float16)o;
        }
}

// ---------------------------------------------------------------------------
// Output projection: [MTOT,DIN] @ [DIN,DIN] + bias -> f32 out.
// Wave tile 32x64 (2x4 WMMA), direct-from-global fragments.
// ---------------------------------------------------------------------------
__global__ void __launch_bounds__(256) k_out_gemm(const _Float16* __restrict__ Cx,
                                                  const _Float16* __restrict__ Wt,
                                                  const float* __restrict__ bias,
                                                  float* __restrict__ out) {
    const int lane = threadIdx.x & 31;
    const int wave = threadIdx.x >> 5;
    const int m0 = blockIdx.x * 64 + (wave & 1) * 32;
    const int n0 = blockIdx.y * 256 + (wave >> 1) * 64;
    v8f acc[2][4] = {};
    for (int k0 = 0; k0 < DIN; k0 += 32) {
        v16h a0 = load_frag_a(Cx + (size_t)m0 * DIN + k0, DIN, lane);
        v16h a1 = load_frag_a(Cx + (size_t)(m0 + 16) * DIN + k0, DIN, lane);
        for (int j = 0; j < 4; ++j) {
            v16h b = load_frag_b(Wt + (size_t)(n0 + j * 16) * DIN + k0, DIN, lane);
            acc[0][j] = wmma_f16(a0, b, acc[0][j]);
            acc[1][j] = wmma_f16(a1, b, acc[1][j]);
        }
    }
    const int nn = lane & 15, hf = lane >> 4;
    for (int i = 0; i < 2; ++i)
        for (int j = 0; j < 4; ++j)
            for (int r = 0; r < 8; ++r) {
                int row = m0 + i * 16 + r + 8 * hf;
                int col = n0 + j * 16 + nn;
                out[(size_t)row * DIN + col] = acc[i][j][r] + bias[col];
            }
}

// ---------------------------------------------------------------------------
extern "C" void kernel_launch(void* const* d_in, const int* in_sizes, int n_in,
                              void* d_out, int out_size, void* d_ws, size_t ws_size,
                              hipStream_t stream) {
    (void)in_sizes; (void)n_in; (void)out_size; (void)ws_size;
    const float* x    = (const float*)d_in[0];  // [2,2048,1024]
    const float* Wqkv = (const float*)d_in[1];  // [1024,3072]
    const float* bqkv = (const float*)d_in[2];  // [3072]
    const float* Wout = (const float*)d_in[3];  // [1024,1024]
    const float* bout = (const float*)d_in[4];  // [1024]
    float* out = (float*)d_out;                 // [2,2048,1024] f32

    // Workspace layout (f16), ~48 MiB total
    _Float16* xh    = (_Float16*)d_ws;                       // MTOT*DIN
    _Float16* wqkvT = xh    + (size_t)MTOT * DIN;            // DQKV*DIN (transposed)
    _Float16* woutT = wqkvT + (size_t)DQKV * DIN;            // DIN*DIN (transposed)
    _Float16* Qh    = woutT + (size_t)DIN * DIN;             // BHN*SEQ*HDIM
    _Float16* Kh    = Qh    + (size_t)BHN * SEQ * HDIM;
    _Float16* Vt    = Kh    + (size_t)BHN * SEQ * HDIM;      // [bh][hd][t]
    _Float16* ctxh  = Vt    + (size_t)BHN * SEQ * HDIM;      // MTOT*DIN

    k_cvt<<<(MTOT * DIN + 255) / 256, 256, 0, stream>>>(x, xh, MTOT * DIN);
    k_cvt_t<<<(DIN * DQKV + 255) / 256, 256, 0, stream>>>(Wqkv, wqkvT, DIN, DQKV);
    k_cvt_t<<<(DIN * DIN + 255) / 256, 256, 0, stream>>>(Wout, woutT, DIN, DIN);

    k_qkv_gemm<<<dim3(MTOT / 64, DQKV / 256), 256, 0, stream>>>(xh, wqkvT, bqkv, Qh, Kh, Vt);
    k_attn<<<dim3(SEQ / 128, BHN), 256, 0, stream>>>(Qh, Kh, Vt, ctxh);
    k_out_gemm<<<dim3(MTOT / 64, DIN / 256), 256, 0, stream>>>(ctxh, woutT, bout, out);
}
